// Qwen3OmniTalker_37520834298110
// MI455X (gfx1250) — matrix operand order repaired and checked by reading, not compiled, not run on hardware
//
#include <hip/hip_runtime.h>
#include <hip/hip_bf16.h>

#define H_DIM   2048
#define I_DIM   1408
#define IS_DIM  5632
#define E_DIM   32
#define K_TOP   4
#define T_DIM   2048

typedef __attribute__((ext_vector_type(16))) __bf16 v16bf;
typedef __attribute__((ext_vector_type(8)))  float  v8f;
typedef __attribute__((ext_vector_type(4)))  unsigned int v4u;
typedef __attribute__((ext_vector_type(8)))  int v8i_t;
typedef __attribute__((ext_vector_type(4)))  int v4i_t;

__device__ __forceinline__ v8f wmma_bf16(v16bf a, v16bf b, v8f c) {
    return __builtin_amdgcn_wmma_f32_16x16x32_bf16(
        false, a, false, b, (short)0, c, false, false);
}

// A-matrix 16x32 bf16 fragment from packed bf16 (K pairs consecutive -> b32 loads).
// ISA 7.12.2: lanes 0-15 hold K={0..7,16..23}, lanes 16-31 hold K={8..15,24..31}.
__device__ __forceinline__ v16bf load_a_bf16(const __bf16* row, int kb, int lane) {
    int klo = (lane & 16) ? 8 : 0;
    v16bf a;
#pragma unroll
    for (int v = 0; v < 4; ++v) {
        int k = kb + klo + 2 * v;
        a[2 * v]     = row[k];
        a[2 * v + 1] = row[k + 1];
    }
#pragma unroll
    for (int v = 0; v < 4; ++v) {
        int k = kb + 16 + klo + 2 * v;
        a[8 + 2 * v]     = row[k];
        a[8 + 2 * v + 1] = row[k + 1];
    }
    return a;
}

// B-matrix 32x16 bf16 fragment from fp32 W[k*ldb + n], n = this lane's column.
__device__ __forceinline__ v16bf load_b_f32(const float* __restrict__ W, int ldb,
                                            int n, int kb, int lane) {
    int kbase = kb + ((lane & 16) ? 8 : 0);
    v16bf b;
#pragma unroll
    for (int v = 0; v < 8; ++v) {
        int k = kbase + 2 * v + (v >= 4 ? 8 : 0);
        b[2 * v]     = (__bf16)W[(size_t)k * ldb + n];
        b[2 * v + 1] = (__bf16)W[(size_t)(k + 1) * ldb + n];
    }
    return b;
}

__device__ __forceinline__ float fast_silu(float g) {
    return g * __builtin_amdgcn_rcpf(1.f + __expf(-g));
}
__device__ __forceinline__ float fast_sigmoid(float g) {
    return __builtin_amdgcn_rcpf(1.f + __expf(-g));
}

// ---------------------------------------------------------------------------
// 0) prep: zero output + counters, convert x -> bf16 copy
// ---------------------------------------------------------------------------
__global__ void prep_kernel(const float* __restrict__ x, __bf16* __restrict__ xb,
                            float* __restrict__ out, int n, int* __restrict__ cnt) {
    int i = blockIdx.x * blockDim.x + threadIdx.x;
    if (i < n) { out[i] = 0.f; xb[i] = (__bf16)x[i]; }
    if (i < E_DIM) cnt[i] = 0;
}

// ---------------------------------------------------------------------------
// 1) router: one wave32 per token (lane e == expert e), plus shared sigmoid gate
// ---------------------------------------------------------------------------
__global__ __launch_bounds__(256) void router_kernel(
    const float* __restrict__ x, const float* __restrict__ gate_w,
    const float* __restrict__ shared_gate,
    int* __restrict__ ecnt, int* __restrict__ etok, float* __restrict__ ewt,
    float* __restrict__ gate_s)
{
    int t = blockIdx.x * 8 + (threadIdx.x >> 5);
    int lane = threadIdx.x & 31;
    if (t >= T_DIM) return;
    const float* xr = x + (size_t)t * H_DIM;

    float logit = 0.f, sg = 0.f;
    for (int h = 0; h < H_DIM; ++h)
        logit = fmaf(xr[h], gate_w[h * E_DIM + lane], logit);
    for (int h = lane; h < H_DIM; h += 32)
        sg = fmaf(xr[h], shared_gate[h], sg);
#pragma unroll
    for (int o = 16; o > 0; o >>= 1) sg += __shfl_xor(sg, o, 32);
    if (lane == 0) gate_s[t] = fast_sigmoid(sg);

    float mx = logit;
#pragma unroll
    for (int o = 16; o > 0; o >>= 1) mx = fmaxf(mx, __shfl_xor(mx, o, 32));
    float p = __expf(logit - mx);
    float sum = p;
#pragma unroll
    for (int o = 16; o > 0; o >>= 1) sum += __shfl_xor(sum, o, 32);
    p *= __builtin_amdgcn_rcpf(sum);

    float pv = p, myw = 0.f, wsum = 0.f;
    bool sel = false;
#pragma unroll
    for (int kk = 0; kk < K_TOP; ++kk) {
        float v = pv; int ei = lane;
#pragma unroll
        for (int o = 16; o > 0; o >>= 1) {
            float ov = __shfl_xor(v, o, 32);
            int   oe = __shfl_xor(ei, o, 32);
            if (ov > v || (ov == v && oe < ei)) { v = ov; ei = oe; }
        }
        if (lane == ei) { sel = true; myw = pv; pv = -1.f; }
        wsum += v;
    }
    if (sel) {
        float w = myw * __builtin_amdgcn_rcpf(wsum);
        int slot = atomicAdd(&ecnt[lane], 1);
        etok[lane * T_DIM + slot] = t;
        ewt[lane * T_DIM + slot]  = w;
    }
}

// ---------------------------------------------------------------------------
// 2) routed experts: block = (expert, 16-token tile), 8 waves.
//    TDM gather-mode stages the 16 gathered bf16 x-rows into LDS (64 KB),
//    then bf16 WMMA gate_up -> SiLU*mul in LDS -> WMMA down-proj -> atomic add.
// ---------------------------------------------------------------------------
__global__ __launch_bounds__(256) void moe_routed_kernel(
    const __bf16* __restrict__ xb,        // [T, H] bf16
    const float* __restrict__ w_gate_up,  // [E, H, 2I]
    const float* __restrict__ w_down,     // [E, I, H]
    const int* __restrict__ etok, const float* __restrict__ ewt,
    const int* __restrict__ ecnt,
    float* __restrict__ out)
{
    __shared__ __bf16 s_x[16 * H_DIM];     // 64 KB gathered x tile (TDM dest)
    __shared__ __bf16 s_act[16 * I_DIM];   // 44 KB SwiGLU intermediate
    __shared__ int    s_tok[16];
    __shared__ float  s_wt[16];

    int e  = blockIdx.x >> 7;
    int ti = blockIdx.x & 127;
    int cnt = ecnt[e];
    if (ti * 16 >= cnt) return;

    int tid = threadIdx.x;
    int lane = tid & 31;
    int wid  = tid >> 5;

    if (tid < 16) {
        int r  = ti * 16 + tid;
        int rc = (r < cnt) ? r : (cnt - 1);
        s_tok[tid] = etok[e * T_DIM + rc];
        s_wt[tid]  = (r < cnt) ? ewt[e * T_DIM + rc] : 0.f;
    }
    __syncthreads();

    // ---- TDM gather: 16 token rows of xb -> s_x (one wave issues the DMA)
    if (wid == 0) {
        unsigned long long ga = (unsigned long long)(uintptr_t)xb;
        unsigned int ldsa = (unsigned int)(uintptr_t)&s_x[0];
        v4u g0;
        g0[0] = 0x80000001u;                        // count=1 | gather_mode | 16b idx
        g0[1] = ldsa;                               // lds_addr
        g0[2] = (unsigned int)(ga & 0xFFFFFFFFu);   // global_addr lo
        g0[3] = (unsigned int)((ga >> 32) & 0x01FFFFFFu) | 0x80000000u; // hi | type=2
        v8i_t g1;
        g1[0] = 0x00010000;                         // data_size=1 (2 bytes)
        g1[1] = (int)((H_DIM & 0xFFFF) << 16);      // tensor_dim0 lo16
        g1[2] = (int)((T_DIM & 0xFFFF) << 16);      // dim0 hi | tensor_dim1 lo16
        g1[3] = (int)((H_DIM & 0xFFFF) << 16);      // dim1 hi | tile_dim0 (row width)
        g1[4] = 16;                                 // tile_dim1 = #valid indices
        g1[5] = H_DIM;                              // tensor_dim0_stride lo32
        g1[6] = 0;
        g1[7] = 0;
        v4i_t g2, g3;                               // 16-bit row indices
#pragma unroll
        for (int i = 0; i < 4; ++i) {
            g2[i] = __builtin_amdgcn_readfirstlane(
                (s_tok[2 * i] & 0xFFFF) | (s_tok[2 * i + 1] << 16));
            g3[i] = __builtin_amdgcn_readfirstlane(
                (s_tok[8 + 2 * i] & 0xFFFF) | (s_tok[8 + 2 * i + 1] << 16));
        }
        v8i_t gx = {};                              // unused 5th group (6-arg form)
        __builtin_amdgcn_tensor_load_to_lds(g0, g1, g2, g3, gx, 0);
        __builtin_amdgcn_s_wait_tensorcnt(0);
    }
    __syncthreads();

    int m = lane & 15;
    const __bf16* axrow = &s_x[m * H_DIM];
    const float* Wgu = w_gate_up + (size_t)e * H_DIM * (2 * I_DIM);

    // ---- phase A: gate_up GEMM + SiLU*mul -> LDS (88 N-tiles over 8 waves)
    for (int jt = wid; jt < I_DIM / 16; jt += 8) {
        int ng = jt * 16 + m;
        v8f cg = {}; v8f cu = {};
        for (int kb = 0; kb < H_DIM; kb += 32) {
            __builtin_prefetch(&Wgu[(size_t)(kb + 32) * (2 * I_DIM) + ng], 0, 0);
            v16bf a  = load_a_bf16(axrow, kb, lane);
            v16bf bg = load_b_f32(Wgu, 2 * I_DIM, ng, kb, lane);
            v16bf bu = load_b_f32(Wgu, 2 * I_DIM, I_DIM + ng, kb, lane);
            cg = wmma_bf16(a, bg, cg);
            cu = wmma_bf16(a, bu, cu);
        }
#pragma unroll
        for (int r = 0; r < 8; ++r) {
            int row = r + ((lane & 16) ? 8 : 0);
            s_act[row * I_DIM + ng] = (__bf16)(fast_silu(cg[r]) * cu[r]);
        }
    }
    __syncthreads();

    // ---- phase B: down-proj from LDS, weighted atomic add to out
    const float* Wd = w_down + (size_t)e * I_DIM * H_DIM;
    const __bf16* arow = &s_act[m * I_DIM];
    for (int nt = wid; nt < H_DIM / 16; nt += 8) {
        int n = nt * 16 + m;
        v8f c = {};
        for (int kb = 0; kb < I_DIM; kb += 32) {
            __builtin_prefetch(&Wd[(size_t)(kb + 32) * H_DIM + n], 0, 0);
            v16bf a = load_a_bf16(arow, kb, lane);
            v16bf b = load_b_f32(Wd, H_DIM, n, kb, lane);
            c = wmma_bf16(a, b, c);
        }
#pragma unroll
        for (int r = 0; r < 8; ++r) {
            int row = r + ((lane & 16) ? 8 : 0);
            atomicAdd(&out[(size_t)s_tok[row] * H_DIM + n], c[r] * s_wt[row]);
        }
    }
}

// ---------------------------------------------------------------------------
// 3) shared expert pass 1: M=32 token tile (2 WMMA row-tiles reuse each B frag)
// ---------------------------------------------------------------------------
__global__ __launch_bounds__(256) void shared_up_kernel(
    const __bf16* __restrict__ xb, const float* __restrict__ sgu,  // [H, 2*IS]
    __bf16* __restrict__ mid)                                      // [T, IS]
{
    int t0 = blockIdx.x * 32;
    int tid = threadIdx.x, lane = tid & 31, wid = tid >> 5;
    int m = lane & 15;
    const __bf16* ar0 = xb + (size_t)(t0 + m) * H_DIM;
    const __bf16* ar1 = xb + (size_t)(t0 + 16 + m) * H_DIM;

    for (int jt = wid; jt < IS_DIM / 16; jt += 8) {
        int ng = jt * 16 + m;
        v8f cg0 = {}, cu0 = {}, cg1 = {}, cu1 = {};
        for (int kb = 0; kb < H_DIM; kb += 32) {
            __builtin_prefetch(&sgu[(size_t)(kb + 32) * (2 * IS_DIM) + ng], 0, 0);
            v16bf a0 = load_a_bf16(ar0, kb, lane);
            v16bf a1 = load_a_bf16(ar1, kb, lane);
            v16bf bg = load_b_f32(sgu, 2 * IS_DIM, ng, kb, lane);
            v16bf bu = load_b_f32(sgu, 2 * IS_DIM, IS_DIM + ng, kb, lane);
            cg0 = wmma_bf16(a0, bg, cg0);
            cu0 = wmma_bf16(a0, bu, cu0);
            cg1 = wmma_bf16(a1, bg, cg1);
            cu1 = wmma_bf16(a1, bu, cu1);
        }
#pragma unroll
        for (int r = 0; r < 8; ++r) {
            int row = r + ((lane & 16) ? 8 : 0);
            mid[(size_t)(t0 + row) * IS_DIM + ng]      = (__bf16)(fast_silu(cg0[r]) * cu0[r]);
            mid[(size_t)(t0 + 16 + row) * IS_DIM + ng] = (__bf16)(fast_silu(cg1[r]) * cu1[r]);
        }
    }
}

// ---------------------------------------------------------------------------
// 4) shared expert pass 2: M=32 down-proj from bf16 scratch, sigmoid-gated add
// ---------------------------------------------------------------------------
__global__ __launch_bounds__(256) void shared_down_kernel(
    const __bf16* __restrict__ mid, const float* __restrict__ sdown,  // [IS, H]
    const float* __restrict__ gate_s, float* __restrict__ out)
{
    int t0 = blockIdx.x * 32;
    int tid = threadIdx.x, lane = tid & 31, wid = tid >> 5;
    int m = lane & 15;
    const __bf16* ar0 = mid + (size_t)(t0 + m) * IS_DIM;
    const __bf16* ar1 = mid + (size_t)(t0 + 16 + m) * IS_DIM;

    for (int nt = wid; nt < H_DIM / 16; nt += 8) {
        int n = nt * 16 + m;
        v8f c0 = {}, c1 = {};
        for (int kb = 0; kb < IS_DIM; kb += 32) {
            __builtin_prefetch(&sdown[(size_t)(kb + 32) * H_DIM + n], 0, 0);
            v16bf a0 = load_a_bf16(ar0, kb, lane);
            v16bf a1 = load_a_bf16(ar1, kb, lane);
            v16bf b  = load_b_f32(sdown, H_DIM, n, kb, lane);
            c0 = wmma_bf16(a0, b, c0);
            c1 = wmma_bf16(a1, b, c1);
        }
#pragma unroll
        for (int r = 0; r < 8; ++r) {
            int row = r + ((lane & 16) ? 8 : 0);
            int ta = t0 + row, tb = t0 + 16 + row;
            atomicAdd(&out[(size_t)ta * H_DIM + n], c0[r] * gate_s[ta]);
            atomicAdd(&out[(size_t)tb * H_DIM + n], c1[r] * gate_s[tb]);
        }
    }
}

// ---------------------------------------------------------------------------
extern "C" void kernel_launch(void* const* d_in, const int* in_sizes, int n_in,
                              void* d_out, int out_size, void* d_ws, size_t ws_size,
                              hipStream_t stream) {
    const float* x       = (const float*)d_in[0];  // [T, H]
    const float* gate_w  = (const float*)d_in[1];  // [H, E]
    const float* w_gu    = (const float*)d_in[2];  // [E, H, 2I]
    const float* w_dn    = (const float*)d_in[3];  // [E, I, H]
    const float* sgu     = (const float*)d_in[4];  // [H, 2*IS]
    const float* sdown   = (const float*)d_in[5];  // [IS, H]
    const float* sgate   = (const float*)d_in[6];  // [H, 1]
    float* out = (float*)d_out;

    // workspace layout
    char* ws = (char*)d_ws;
    size_t off = 0;
    __bf16* xb  = (__bf16*)(ws + off); off += (size_t)T_DIM * H_DIM * sizeof(__bf16);   // 8 MB
    __bf16* mid = (__bf16*)(ws + off); off += (size_t)T_DIM * IS_DIM * sizeof(__bf16);  // 22 MB
    int*    etok = (int*)(ws + off);   off += (size_t)E_DIM * T_DIM * sizeof(int);
    float*  ewt  = (float*)(ws + off); off += (size_t)E_DIM * T_DIM * sizeof(float);
    int*    ecnt = (int*)(ws + off);   off += 128;
    float*  gs   = (float*)(ws + off); off += (size_t)T_DIM * sizeof(float);

    int n = T_DIM * H_DIM;
    prep_kernel<<<(n + 255) / 256, 256, 0, stream>>>(x, xb, out, n, ecnt);

    router_kernel<<<T_DIM / 8, 256, 0, stream>>>(x, gate_w, sgate, ecnt, etok, ewt, gs);

    moe_routed_kernel<<<E_DIM * (T_DIM / 16), 256, 0, stream>>>(
        xb, w_gu, w_dn, etok, ewt, ecnt, out);

    shared_up_kernel<<<T_DIM / 32, 256, 0, stream>>>(xb, sgu, mid);

    shared_down_kernel<<<T_DIM / 32, 256, 0, stream>>>(mid, sdown, gs, out);
}
